// BrainRNN_37684043055467
// MI455X (gfx1250) — matrix operand order repaired and checked by reading, not compiled, not run on hardware
//
#include <hip/hip_runtime.h>
#include <hip/hip_bf16.h>

typedef __attribute__((ext_vector_type(16))) __bf16 bf16x16;
typedef __attribute__((ext_vector_type(8)))  float  f32x8;
typedef __attribute__((ext_vector_type(4)))  float  f32x4;

#define N_NEUR 8192
#define WIDTH  1024
#define NL     8
#define BATCH  64
#define IN_D   512
#define OUT_D  512
#define SKIPW  7168   // (NL-1)*WIDTH

// ---------------------------------------------------------------------------
// Fragment loaders (CDNA5 wave32 WMMA 16x16x32 bf16; contiguous k order used here, same for A and B)
// A 16x32 (MxK): lane L holds row M = L&15; K run [kh*16 .. kh*16+15] in v[0..15],
//                kh = L>>4.
// B 32x16 (KxN): lane L holds col N = L&15; K run [kh*16 .. kh*16+15].
// ---------------------------------------------------------------------------
__device__ __forceinline__ bf16x16 load_a_frag(const float* __restrict__ arow,
                                               int kb, int kh) {
  const float* p = arow + kb + kh * 16;
  f32x4 x0 = *(const f32x4*)(p);
  f32x4 x1 = *(const f32x4*)(p + 4);
  f32x4 x2 = *(const f32x4*)(p + 8);
  f32x4 x3 = *(const f32x4*)(p + 12);
  bf16x16 a;
#pragma unroll
  for (int e = 0; e < 4; ++e) {
    a[e]      = (__bf16)x0[e];
    a[4 + e]  = (__bf16)x1[e];
    a[8 + e]  = (__bf16)x2[e];
    a[12 + e] = (__bf16)x3[e];
  }
  return a;
}

// Weights are single-use: non-temporal loads keep L2 free for activations.
__device__ __forceinline__ bf16x16 load_b_masked(const float* __restrict__ wrow,
                                                 const float* __restrict__ mrow,
                                                 int kb, int kh) {
  const float* pw = wrow + kb + kh * 16;
  const float* pm = mrow + kb + kh * 16;
  bf16x16 b;
#pragma unroll
  for (int q = 0; q < 4; ++q) {
    f32x4 w = __builtin_nontemporal_load((const f32x4*)(pw + 4 * q));
    f32x4 m = __builtin_nontemporal_load((const f32x4*)(pm + 4 * q));
#pragma unroll
    for (int e = 0; e < 4; ++e) b[4 * q + e] = (__bf16)(w[e] * m[e]);
  }
  return b;
}

__device__ __forceinline__ bf16x16 load_b_dense(const float* __restrict__ wrow,
                                                int kb, int kh) {
  const float* pw = wrow + kb + kh * 16;
  bf16x16 b;
#pragma unroll
  for (int q = 0; q < 4; ++q) {
    f32x4 w = __builtin_nontemporal_load((const f32x4*)(pw + 4 * q));
#pragma unroll
    for (int e = 0; e < 4; ++e) b[4 * q + e] = (__bf16)w[e];
  }
  return b;
}

__device__ __forceinline__ f32x8 wmma_bf16(bf16x16 a, bf16x16 b, f32x8 c) {
  return __builtin_amdgcn_wmma_f32_16x16x32_bf16(false, a, false, b,
                                                 (short)0, c, false, false);
}

__device__ __forceinline__ float sigmoidf_fast(float x) {
  return 1.0f / (1.0f + __expf(-x));
}

// ---------------------------------------------------------------------------
// Kernel 1: R[l] = h @ (W_rec[l]*mask_rec[l])^T  for l = 0..6  (fully parallel)
// 1792 waves: (layer, mtile, ntile); one 16x16 tile per wave over K=8192.
// ---------------------------------------------------------------------------
__global__ void __launch_bounds__(256)
brainrnn_rec_gemm(const float* __restrict__ h,     // [64, 8192]
                  const float* __restrict__ Wrec,  // [7, 1024, 8192]
                  const float* __restrict__ Mrec,  // [7, 1024, 8192]
                  float* __restrict__ R) {         // [7, 64, 1024]
  const int lane  = threadIdx.x & 31;
  const int wid   = blockIdx.x * (blockDim.x >> 5) + (threadIdx.x >> 5);
  const int layer = wid >> 8;        // 256 waves per layer
  const int rem   = wid & 255;
  const int mtile = rem >> 6;        // 0..3
  const int ntile = rem & 63;        // 0..63
  const int kh    = lane >> 4;
  const int nloc  = lane & 15;

  const float* arow = h + (size_t)(mtile * 16 + nloc) * N_NEUR;  // row M = lane&15
  const float* wrow = Wrec + ((size_t)layer * WIDTH + ntile * 16 + nloc) * N_NEUR;
  const float* mrow = Mrec + ((size_t)layer * WIDTH + ntile * 16 + nloc) * N_NEUR;

  f32x8 acc = {};
#pragma unroll 2
  for (int kb = 0; kb < N_NEUR; kb += 32) {
    bf16x16 a = load_a_frag(arow, kb, kh);
    bf16x16 b = load_b_masked(wrow, mrow, kb, kh);
    acc = wmma_bf16(a, b, acc);
  }
  float* out = R + ((size_t)layer * BATCH + mtile * 16 + kh * 8) * WIDTH +
               ntile * 16 + nloc;
#pragma unroll
  for (int v = 0; v < 8; ++v) out[(size_t)v * WIDTH] = acc[v];
}

// ---------------------------------------------------------------------------
// Kernel 2: skips[:, 0:1024] = sigmoid(x @ W_in^T + b_in + R[0])
// 256 waves: (mtile, ntile), K = 512 dense.
// ---------------------------------------------------------------------------
__global__ void __launch_bounds__(256)
brainrnn_input_layer(const float* __restrict__ x,    // [64, 512]
                     const float* __restrict__ Win,  // [1024, 512]
                     const float* __restrict__ bin,  // [1024]
                     const float* __restrict__ R,    // [7, 64, 1024]
                     float* __restrict__ skips) {    // [64, 8192]
  const int lane  = threadIdx.x & 31;
  const int wid   = blockIdx.x * (blockDim.x >> 5) + (threadIdx.x >> 5);
  const int mtile = wid >> 6;
  const int ntile = wid & 63;
  const int kh    = lane >> 4;
  const int nloc  = lane & 15;

  const float* arow = x + (size_t)(mtile * 16 + nloc) * IN_D;
  const float* wrow = Win + (size_t)(ntile * 16 + nloc) * IN_D;

  f32x8 acc = {};
#pragma unroll 2
  for (int kb = 0; kb < IN_D; kb += 32) {
    bf16x16 a = load_a_frag(arow, kb, kh);
    bf16x16 b = load_b_dense(wrow, kb, kh);
    acc = wmma_bf16(a, b, acc);
  }
  const int ncol = ntile * 16 + nloc;
  const float bv = bin[ncol];
#pragma unroll
  for (int v = 0; v < 8; ++v) {
    const int mrow = mtile * 16 + kh * 8 + v;
    float y = acc[v] + bv + R[(size_t)mrow * WIDTH + ncol];  // R[0]
    skips[(size_t)mrow * N_NEUR + ncol] = sigmoidf_fast(y);
  }
}

// ---------------------------------------------------------------------------
// Kernel 3 (x7, stream-ordered): one chain step.
// y = cur_i @ (Whid[i]*Mhid[i])^T + b_hid[i]
//     + (i>=1) sum_{j<=i} skips_j @ (Wskip[i-1]*Mskip[i-1])_j^T   (triangular)
//     + (i<6)  R[i+1]
// skips[:, (i+1)w] = sigmoid(y)
// ---------------------------------------------------------------------------
__global__ void __launch_bounds__(256)
brainrnn_layer(const float* __restrict__ Whid,   // [7, 1024, 1024]
               const float* __restrict__ Mhid,
               const float* __restrict__ bhid,   // [7, 1024]
               const float* __restrict__ Wskip,  // [6, 1024, 7168]
               const float* __restrict__ Mskip,
               const float* __restrict__ R,      // [7, 64, 1024]
               float* __restrict__ skips,        // [64, 8192]
               int step) {                       // 0..6
  const int lane  = threadIdx.x & 31;
  const int wid   = blockIdx.x * (blockDim.x >> 5) + (threadIdx.x >> 5);
  const int mtile = wid >> 6;
  const int ntile = wid & 63;
  const int kh    = lane >> 4;
  const int nloc  = lane & 15;

  const size_t arow_base = (size_t)(mtile * 16 + nloc) * N_NEUR;
  f32x8 acc = {};

  // hidden term: A = skips block `step`, W = Whid[step]
  {
    const float* arow = skips + arow_base + (size_t)step * WIDTH;
    const float* wrow = Whid + ((size_t)step * WIDTH + ntile * 16 + nloc) * WIDTH;
    const float* mrow = Mhid + ((size_t)step * WIDTH + ntile * 16 + nloc) * WIDTH;
#pragma unroll 2
    for (int kb = 0; kb < WIDTH; kb += 32) {
      bf16x16 a = load_a_frag(arow, kb, kh);
      bf16x16 b = load_b_masked(wrow, mrow, kb, kh);
      acc = wmma_bf16(a, b, acc);
    }
  }

  // skip term: only the (step+1)*1024 columns where mask_skip can be nonzero
  if (step >= 1) {
    const int kmax = (step + 1) * WIDTH;
    const float* arow = skips + arow_base;
    const float* wrow = Wskip + ((size_t)(step - 1) * WIDTH + ntile * 16 + nloc) * SKIPW;
    const float* mrow = Mskip + ((size_t)(step - 1) * WIDTH + ntile * 16 + nloc) * SKIPW;
#pragma unroll 2
    for (int kb = 0; kb < kmax; kb += 32) {
      bf16x16 a = load_a_frag(arow, kb, kh);
      bf16x16 b = load_b_masked(wrow, mrow, kb, kh);
      acc = wmma_bf16(a, b, acc);
    }
  }

  const int ncol = ntile * 16 + nloc;
  const float bv = bhid[(size_t)step * WIDTH + ncol];
#pragma unroll
  for (int v = 0; v < 8; ++v) {
    const int mrow = mtile * 16 + kh * 8 + v;
    float y = acc[v] + bv;
    if (step < NL - 2)
      y += R[((size_t)(step + 1) * BATCH + mrow) * WIDTH + ncol];
    skips[(size_t)mrow * N_NEUR + (size_t)(step + 1) * WIDTH + ncol] =
        sigmoidf_fast(y);
  }
}

// ---------------------------------------------------------------------------
// Kernel 4: out = cur7 @ W_out^T + b_out   (cur7 = skips block 7)
// 128 waves: 4 mtiles x 32 ntiles, K = 1024 dense.
// ---------------------------------------------------------------------------
__global__ void __launch_bounds__(256)
brainrnn_out(const float* __restrict__ Wout,   // [512, 1024]
             const float* __restrict__ bout,   // [512]
             const float* __restrict__ skips,  // [64, 8192]
             float* __restrict__ out) {        // [64, 512]
  const int lane  = threadIdx.x & 31;
  const int wid   = blockIdx.x * (blockDim.x >> 5) + (threadIdx.x >> 5);
  const int mtile = wid >> 5;        // 0..3
  const int ntile = wid & 31;        // 0..31
  const int kh    = lane >> 4;
  const int nloc  = lane & 15;

  const float* arow = skips + (size_t)(mtile * 16 + nloc) * N_NEUR + 7 * WIDTH;
  const float* wrow = Wout + (size_t)(ntile * 16 + nloc) * WIDTH;

  f32x8 acc = {};
#pragma unroll 2
  for (int kb = 0; kb < WIDTH; kb += 32) {
    bf16x16 a = load_a_frag(arow, kb, kh);
    bf16x16 b = load_b_dense(wrow, kb, kh);
    acc = wmma_bf16(a, b, acc);
  }
  const int ncol = ntile * 16 + nloc;
  const float bv = bout[ncol];
#pragma unroll
  for (int v = 0; v < 8; ++v) {
    const int mrow = mtile * 16 + kh * 8 + v;
    out[(size_t)mrow * OUT_D + ncol] = acc[v] + bv;
  }
}

// ---------------------------------------------------------------------------
extern "C" void kernel_launch(void* const* d_in, const int* in_sizes, int n_in,
                              void* d_out, int out_size, void* d_ws, size_t ws_size,
                              hipStream_t stream) {
  const float* x     = (const float*)d_in[0];
  const float* h     = (const float*)d_in[1];
  const float* Win   = (const float*)d_in[2];
  const float* bin   = (const float*)d_in[3];
  const float* Whid  = (const float*)d_in[4];
  const float* bhid  = (const float*)d_in[5];
  const float* Wrec  = (const float*)d_in[6];
  const float* Wskip = (const float*)d_in[7];
  const float* Wout  = (const float*)d_in[8];
  const float* bout  = (const float*)d_in[9];
  const float* Mhid  = (const float*)d_in[10];
  const float* Mrec  = (const float*)d_in[11];
  const float* Mskip = (const float*)d_in[12];
  float* out = (float*)d_out;

  // workspace: R [7,64,1024] then skips [64,8192]; both fully written before read
  float* R     = (float*)d_ws;
  float* skips = R + (size_t)(NL - 1) * BATCH * WIDTH;

  // Phase 1: all recurrent terms, fully parallel (dominant 460 MB stream)
  // 1792 waves = 224 blocks x 8 waves
  brainrnn_rec_gemm<<<224, 256, 0, stream>>>(h, Wrec, Mrec, R);

  // Phase 2: input block (256 waves = 32 blocks)
  brainrnn_input_layer<<<32, 256, 0, stream>>>(x, Win, bin, R, skips);

  // Phase 3: sequential chain (stream order gives the dependency)
  for (int i = 0; i < NL - 1; ++i)
    brainrnn_layer<<<32, 256, 0, stream>>>(Whid, Mhid, bhid, Wskip, Mskip,
                                           R, skips, i);

  // Phase 4: output projection (128 waves = 16 blocks)
  brainrnn_out<<<16, 256, 0, stream>>>(Wout, bout, skips, out);
}
